// GraphTransformerAutoEncoder_50371376447825
// MI455X (gfx1250) — compile-verified, matrix operand order checked by reading, not run on hardware
//
#include <hip/hip_runtime.h>
#include <math.h>

// ---------------- constants (match reference) ----------------
constexpr int cN0 = 10000, cN1 = 5000, cN2 = 2500;
constexpr int cE  = 160000;
constexpr int WG  = 256;

typedef __bf16 bf16_t;
typedef __attribute__((ext_vector_type(16))) __bf16 v16bf;
typedef __attribute__((ext_vector_type(8)))  float  v8f;
typedef __attribute__((ext_vector_type(4)))  float  v4f;

static inline int cdiv(int a, int b) { return (a + b - 1) / b; }

// ---------------- ordered-float encoding for atomicMax ----------------
__device__ __forceinline__ unsigned fenc(float f) {
  unsigned u = __float_as_uint(f);
  return (u & 0x80000000u) ? ~u : (u | 0x80000000u);
}
__device__ __forceinline__ float fdec(unsigned u) {
  u = (u & 0x80000000u) ? (u & 0x7fffffffu) : ~u;
  return __uint_as_float(u);
}

// ---------------- weight pre-pack into WMMA B-fragment layout ----------------
// fragment f = (nt*(K/32) + kt)*32 + lane holds 16 bf16: B[k, nt*16 + (lane&15)]
// with k = kt*32 + (lane>>4)*8 + (e<8 ? e : 8+e)
__global__ void k_pack_w(const float* __restrict__ W, bf16_t* __restrict__ Wp,
                         int K, int Nc) {
  const int KT = K >> 5;
  const int nfrag = (Nc >> 4) * KT * 32;
  int idx = blockIdx.x * blockDim.x + threadIdx.x;
  if (idx >= nfrag) return;
  const int lane = idx & 31;
  const int t    = idx >> 5;
  const int kt   = t % KT;
  const int nt   = t / KT;
  const int col   = nt * 16 + (lane & 15);
  const int kbase = kt * 32 + (lane >> 4) * 8;
  bf16_t* o = Wp + (size_t)idx * 16;
#pragma unroll
  for (int e = 0; e < 16; ++e) {
    const int kk = kbase + (e < 8 ? e : 8 + e);
    o[e] = (bf16_t)W[(size_t)kk * Nc + col];
  }
}

// ---------------- WMMA GEMM: C[M,Nc] = A[M,K] * W[K,Nc] + bias ----------------
// One wave -> one 16x32 tile (2 accumulators sharing the A fragment).
// A: row index clamped to M-1 (garbage rows never contaminate stored rows),
//    loaded as 4x b128 per K-step. B: pre-packed bf16 fragments, 1x 32B load each.
// Requires K % 32 == 0, Nc % 32 == 0 (true for all projections here).
__global__ void gemm_bias_wmma(const float* __restrict__ A, const bf16_t* __restrict__ Wp,
                               const float* __restrict__ bias, float* __restrict__ C,
                               int M, int K, int Nc) {
  const int lane  = threadIdx.x & 31;
  const int wave  = threadIdx.x >> 5;
  const int tileM = (blockIdx.x * (blockDim.x >> 5) + wave) * 16;
  const int nt0   = blockIdx.y * 2;           // two 16-col tiles per wave
  const int half  = lane >> 4;
  const int rA    = tileM + (lane & 15);
  const int col0  = nt0 * 16 + (lane & 15);
  const int kbase = half * 8;
  const int KT    = K >> 5;

  const int rClamp = (rA < M) ? rA : (M - 1);
  const float* __restrict__ Ar = A + (size_t)rClamp * K + kbase;
  const v16bf* __restrict__ Wv = (const v16bf*)Wp;
  const size_t fb0 = ((size_t)nt0 * KT) * 32 + lane;
  const size_t fb1 = ((size_t)(nt0 + 1) * KT) * 32 + lane;

  v8f acc0 = {}, acc1 = {};
  for (int kt = 0; kt < KT; ++kt) {
    const v4f f0 = *(const v4f*)(Ar + kt * 32);
    const v4f f1 = *(const v4f*)(Ar + kt * 32 + 4);
    const v4f f2 = *(const v4f*)(Ar + kt * 32 + 16);
    const v4f f3 = *(const v4f*)(Ar + kt * 32 + 20);
    v16bf a;
#pragma unroll
    for (int e = 0; e < 4; ++e) {
      a[e]      = (bf16_t)f0[e];
      a[4 + e]  = (bf16_t)f1[e];
      a[8 + e]  = (bf16_t)f2[e];
      a[12 + e] = (bf16_t)f3[e];
    }
    const v16bf b0 = Wv[fb0 + (size_t)kt * 32];
    const v16bf b1 = Wv[fb1 + (size_t)kt * 32];
    acc0 = __builtin_amdgcn_wmma_f32_16x16x32_bf16(false, a, false, b0, (short)0, acc0,
                                                   false, false);
    acc1 = __builtin_amdgcn_wmma_f32_16x16x32_bf16(false, a, false, b1, (short)0, acc1,
                                                   false, false);
  }
  const float bv0 = bias ? bias[col0] : 0.0f;
  const float bv1 = bias ? bias[col0 + 16] : 0.0f;
#pragma unroll
  for (int r = 0; r < 8; ++r) {
    const int ro = tileM + r + half * 8;
    if (ro < M) {
      C[(size_t)ro * Nc + col0]      = acc0[r] + bv0;
      C[(size_t)ro * Nc + col0 + 16] = acc1[r] + bv1;
    }
  }
}

// ---------------- utility kernels ----------------
__global__ void k_fill_f32(float* p, float v, int n) {
  int i = blockIdx.x * blockDim.x + threadIdx.x;
  if (i < n) p[i] = v;
}
__global__ void k_fill_u32(unsigned* p, unsigned v, int n) {
  int i = blockIdx.x * blockDim.x + threadIdx.x;
  if (i < n) p[i] = v;
}
__global__ void k_copy_f32(float* d, const float* s, int n) {
  int i = blockIdx.x * blockDim.x + threadIdx.x;
  if (i < n) d[i] = s[i];
}

// ---------------- attention: alpha + segment max ----------------
__global__ void k_alpha_max(const float* __restrict__ q, const float* __restrict__ kk,
                            const float* __restrict__ edge_attr, const float* __restrict__ eW,
                            const int* __restrict__ src, const int* __restrict__ dst,
                            const int* __restrict__ mask,
                            float* __restrict__ alpha, unsigned* __restrict__ mEnc,
                            int E, int H, int Ch, float rsC) {
  int idx = blockIdx.x * blockDim.x + threadIdx.x;
  if (idx >= E * H) return;
  int e = idx / H, h = idx - e * H;
  bool ok = (mask == nullptr) || (mask[e] != 0);
  if (!ok) { alpha[idx] = -1e30f; return; }
  int s = src[e], d = dst[e];
  int hc = H * Ch;
  float ea[16];
#pragma unroll
  for (int t = 0; t < 16; ++t) ea[t] = edge_attr[(long)e * 16 + t];
  float acc = 0.f;
  for (int c = 0; c < Ch; ++c) {
    float ev = 0.f;
#pragma unroll
    for (int t = 0; t < 16; ++t) ev += ea[t] * eW[t * hc + h * Ch + c];
    acc += q[(long)d * hc + h * Ch + c] * (kk[(long)s * hc + h * Ch + c] + ev);
  }
  float a = acc * rsC;
  alpha[idx] = a;
  atomicMax(&mEnc[d * H + h], fenc(a));
}

// ---------------- attention: exp + segment sum ----------------
__global__ void k_exp_den(float* __restrict__ alpha, const unsigned* __restrict__ mEnc,
                          float* __restrict__ den, const int* __restrict__ dst,
                          const int* __restrict__ mask, int E, int H) {
  int idx = blockIdx.x * blockDim.x + threadIdx.x;
  if (idx >= E * H) return;
  int e = idx / H, h = idx - e * H;
  bool ok = (mask == nullptr) || (mask[e] != 0);
  if (!ok) { alpha[idx] = 0.f; return; }
  int d = dst[e];
  float ex = expf(alpha[idx] - fdec(mEnc[d * H + h]));
  alpha[idx] = ex;
  atomicAdd(&den[d * H + h], ex);
}

// ---------------- attention: weighted message scatter-add ----------------
__global__ void k_scatter_msg(const float* __restrict__ v, const float* __restrict__ edge_attr,
                              const float* __restrict__ eW, const float* __restrict__ alpha,
                              const float* __restrict__ den,
                              const int* __restrict__ src, const int* __restrict__ dst,
                              const int* __restrict__ mask, float* __restrict__ agg,
                              int E, int H, int Ch) {
  int hc = H * Ch;
  int idx = blockIdx.x * blockDim.x + threadIdx.x;
  if (idx >= E * hc) return;
  int e = idx / hc, r = idx - e * hc;
  bool ok = (mask == nullptr) || (mask[e] != 0);
  if (!ok) return;
  int h = r / Ch;
  int s = src[e], d = dst[e];
  float al = alpha[e * H + h] / fmaxf(den[d * H + h], 1e-16f);
  float ev = 0.f;
#pragma unroll
  for (int t = 0; t < 16; ++t) ev += edge_attr[(long)e * 16 + t] * eW[t * hc + r];
  atomicAdd(&agg[(long)d * hc + r], (v[(long)s * hc + r] + ev) * al);
}

// ---------------- gated skip (beta) combine ----------------
__global__ void k_beta_out(const float* __restrict__ agg, const float* __restrict__ xr,
                           const float* __restrict__ betaW, float* __restrict__ out,
                           int N, int hc) {
  int n = blockIdx.x * blockDim.x + threadIdx.x;
  if (n >= N) return;
  float bs = 0.f;
  for (int c = 0; c < hc; ++c) {
    float o = agg[(long)n * hc + c], x_r = xr[(long)n * hc + c];
    bs += o * betaW[c] + x_r * betaW[hc + c] + (o - x_r) * betaW[2 * hc + c];
  }
  float be = 1.f / (1.f + expf(-bs));
  for (int c = 0; c < hc; ++c) {
    float o = agg[(long)n * hc + c], x_r = xr[(long)n * hc + c];
    out[(long)n * hc + c] = be * x_r + (1.f - be) * o;
  }
}

// ---------------- batch norm ----------------
__global__ void k_bn_stats(const float* __restrict__ x, float* __restrict__ mean,
                           float* __restrict__ var, int N, int Cch) {
  int ch = blockIdx.x;
  __shared__ float s1[256], s2[256];
  float a = 0.f, b = 0.f;
  for (int n = threadIdx.x; n < N; n += blockDim.x) {
    float v = x[(long)n * Cch + ch]; a += v; b += v * v;
  }
  s1[threadIdx.x] = a; s2[threadIdx.x] = b; __syncthreads();
  for (int o = blockDim.x / 2; o > 0; o >>= 1) {
    if ((int)threadIdx.x < o) { s1[threadIdx.x] += s1[threadIdx.x + o]; s2[threadIdx.x] += s2[threadIdx.x + o]; }
    __syncthreads();
  }
  if (threadIdx.x == 0) {
    float mu = s1[0] / N;
    mean[ch] = mu;
    var[ch] = s2[0] / N - mu * mu;
  }
}
__global__ void k_bn_apply_relu(float* __restrict__ x, const float* __restrict__ mean,
                                const float* __restrict__ var, const float* __restrict__ g,
                                const float* __restrict__ b, int total, int Cch) {
  int i = blockIdx.x * blockDim.x + threadIdx.x;
  if (i >= total) return;
  int ch = i % Cch;
  float y = (x[i] - mean[ch]) * rsqrtf(var[ch] + 1e-5f) * g[ch] + b[ch];
  x[i] = fmaxf(y, 0.f);
}

// ---------------- TopK pooling ----------------
__global__ void k_wnorm(const float* __restrict__ w, float* __restrict__ out) {
  __shared__ float s[128];
  float v = w[threadIdx.x];
  s[threadIdx.x] = v * v; __syncthreads();
  for (int o = 64; o > 0; o >>= 1) {
    if ((int)threadIdx.x < o) s[threadIdx.x] += s[threadIdx.x + o];
    __syncthreads();
  }
  if (threadIdx.x == 0) out[0] = sqrtf(s[0]);
}
__global__ void k_score(const float* __restrict__ x, const float* __restrict__ pw,
                        const float* __restrict__ wn, float* __restrict__ score, int N, int ch) {
  int n = blockIdx.x * blockDim.x + threadIdx.x;
  if (n >= N) return;
  float dot = 0.f;
  for (int c = 0; c < ch; ++c) dot += x[(long)n * ch + c] * pw[c];
  score[n] = tanhf(dot / wn[0]);
}
// exact stable-descending rank (matches jax top_k ordering)
__global__ void k_rank(const float* __restrict__ score, int* __restrict__ rank, int N) {
  int i = blockIdx.x * blockDim.x + threadIdx.x;
  if (i >= N) return;
  float si = score[i];
  int r = 0;
  for (int j = 0; j < N; ++j) {
    float sj = score[j];
    r += (sj > si) || (sj == si && j < i);
  }
  rank[i] = r;
}
__global__ void k_build_perm(const int* __restrict__ rank, int* __restrict__ perm,
                             int* __restrict__ inv, int N, int kkeep) {
  int i = blockIdx.x * blockDim.x + threadIdx.x;
  if (i >= N) return;
  int r = rank[i];
  if (r < kkeep) { perm[r] = i; inv[i] = r; } else inv[i] = -1;
}
__global__ void k_gather_pool(const float* __restrict__ x, const int* __restrict__ perm,
                              const float* __restrict__ score, float* __restrict__ out,
                              int kkeep, int ch) {
  int idx = blockIdx.x * blockDim.x + threadIdx.x;
  if (idx >= kkeep * ch) return;
  int r = idx / ch, c = idx - r * ch;
  int i = perm[r];
  out[idx] = x[(long)i * ch + c] * score[i];
}
__global__ void k_remap(const int* __restrict__ srcO, const int* __restrict__ dstO,
                        const int* __restrict__ maskO, const int* __restrict__ inv,
                        int* __restrict__ nsrc, int* __restrict__ ndst, int* __restrict__ nmask,
                        int E) {
  int e = blockIdx.x * blockDim.x + threadIdx.x;
  if (e >= E) return;
  int ms = (maskO == nullptr) ? 1 : maskO[e];
  int ns = inv[srcO[e]], nd = inv[dstO[e]];
  int nm = (ms != 0) && (ns >= 0) && (nd >= 0);
  nsrc[e] = nm ? ns : 0;
  ndst[e] = nm ? nd : 0;
  nmask[e] = nm;
}

// ---------------- unpool scatter & concat ----------------
__global__ void k_scatter_unpool(const float* __restrict__ x, const int* __restrict__ perm,
                                 float* __restrict__ dstbuf, int kkeep, int ch) {
  int idx = blockIdx.x * blockDim.x + threadIdx.x;
  if (idx >= kkeep * ch) return;
  int r = idx / ch, c = idx - r * ch;
  dstbuf[(long)perm[r] * ch + c] = x[idx];
}
__global__ void k_concat(const float* __restrict__ a, const float* __restrict__ b,
                         float* __restrict__ out, int N, int c1, int c2) {
  int tot = N * (c1 + c2);
  int idx = blockIdx.x * blockDim.x + threadIdx.x;
  if (idx >= tot) return;
  int n = idx / (c1 + c2), c = idx - n * (c1 + c2);
  out[idx] = (c < c1) ? a[(long)n * c1 + c] : b[(long)n * c2 + (c - c1)];
}

// ---------------- host-side helpers ----------------
struct ConvP {
  const float *betaW, *eW, *kW, *kb, *qW, *qb, *skW, *skb, *vW, *vb;
};

static void run_gemm(hipStream_t st, const float* x, const float* W, const float* bias,
                     float* out, int M, int K, int Nc, bf16_t* WPACK) {
  const int nfrag = (Nc / 16) * (K / 32) * 32;
  k_pack_w<<<cdiv(nfrag, WG), WG, 0, st>>>(W, WPACK, K, Nc);
  dim3 blk(128);
  dim3 grd((unsigned)cdiv(M, 64), (unsigned)(Nc / 32));
  gemm_bias_wmma<<<grd, blk, 0, st>>>(x, WPACK, bias, out, M, K, Nc);
}

static void run_conv(hipStream_t st, const ConvP& p, const float* x, int in_ch, int N,
                     const int* src, const int* dst, const int* mask, int H, int Ch,
                     const float* edge_attr, bf16_t* WPACK,
                     float* Q, float* Kb, float* V, float* XR, float* AGG,
                     float* ALPHA, unsigned* MENC, float* DEN, float* out) {
  const int hc = H * Ch;
  run_gemm(st, x, p.qW,  p.qb,  Q,  N, in_ch, hc, WPACK);
  run_gemm(st, x, p.kW,  p.kb,  Kb, N, in_ch, hc, WPACK);
  run_gemm(st, x, p.vW,  p.vb,  V,  N, in_ch, hc, WPACK);
  run_gemm(st, x, p.skW, p.skb, XR, N, in_ch, hc, WPACK);

  unsigned menc0;
  { union { float f; unsigned u; } c; c.f = -1e30f; menc0 = ~c.u; }  // fenc(-1e30)
  k_fill_u32<<<cdiv(N * H, WG), WG, 0, st>>>(MENC, menc0, N * H);
  k_fill_f32<<<cdiv(N * H, WG), WG, 0, st>>>(DEN, 0.f, N * H);
  k_fill_f32<<<cdiv(N * hc, WG), WG, 0, st>>>(AGG, 0.f, N * hc);

  const float rsC = 1.0f / sqrtf((float)Ch);
  k_alpha_max<<<cdiv(cE * H, WG), WG, 0, st>>>(Q, Kb, edge_attr, p.eW, src, dst, mask,
                                               ALPHA, MENC, cE, H, Ch, rsC);
  k_exp_den<<<cdiv(cE * H, WG), WG, 0, st>>>(ALPHA, MENC, DEN, dst, mask, cE, H);
  k_scatter_msg<<<cdiv(cE * hc, WG), WG, 0, st>>>(V, edge_attr, p.eW, ALPHA, DEN,
                                                  src, dst, mask, AGG, cE, H, Ch);
  k_beta_out<<<cdiv(N, WG), WG, 0, st>>>(AGG, XR, p.betaW, out, N, hc);
}

static void run_bn_relu(hipStream_t st, float* x, const float* g, const float* b,
                        int N, int Cch, float* MEAN, float* VARB) {
  k_bn_stats<<<Cch, 256, 0, st>>>(x, MEAN, VARB, N, Cch);
  k_bn_apply_relu<<<cdiv(N * Cch, WG), WG, 0, st>>>(x, MEAN, VARB, g, b, N * Cch, Cch);
}

static void run_pool(hipStream_t st, const float* x, const float* pw, int N, int kkeep,
                     const int* srcO, const int* dstO, const int* maskO,
                     float* SCORE, int* RANK, int* PERM, int* INV, float* WN,
                     float* outX, int* nsrc, int* ndst, int* nmask) {
  k_wnorm<<<1, 128, 0, st>>>(pw, WN);
  k_score<<<cdiv(N, WG), WG, 0, st>>>(x, pw, WN, SCORE, N, 128);
  k_rank<<<cdiv(N, WG), WG, 0, st>>>(SCORE, RANK, N);
  k_build_perm<<<cdiv(N, WG), WG, 0, st>>>(RANK, PERM, INV, N, kkeep);
  k_gather_pool<<<cdiv(kkeep * 128, WG), WG, 0, st>>>(x, PERM, SCORE, outX, kkeep, 128);
  k_remap<<<cdiv(cE, WG), WG, 0, st>>>(srcO, dstO, maskO, INV, nsrc, ndst, nmask, cE);
}

// ---------------- entry point ----------------
extern "C" void kernel_launch(void* const* d_in, const int* in_sizes, int n_in,
                              void* d_out, int out_size, void* d_ws, size_t ws_size,
                              hipStream_t stream) {
  (void)in_sizes; (void)n_in; (void)out_size; (void)ws_size;
  const float* X0 = (const float*)d_in[0];
  const int*   EI = (const int*)d_in[1];
  const float* EA = (const float*)d_in[2];
  const int* SRC0 = EI;
  const int* DST0 = EI + cE;

  auto F = [&](int i) { return (const float*)d_in[i]; };

  // params flattened: top-level insertion order; nested dict keys alphabetical
  // bn_dec[0]{b,g}, bn_dec[1]{b,g}, bn_enc[0..2]{b,g},
  // dec[0..2]{beta.W, e.W, k.W, k.b, q.W, q.b, skip.W, skip.b, v.W, v.b},
  // enc[0..2]{same}, pools[0..1]
  const float* BNdec_b[2] = { F(3), F(5) };
  const float* BNdec_g[2] = { F(4), F(6) };
  const float* BNenc_b[3] = { F(7), F(9), F(11) };
  const float* BNenc_g[3] = { F(8), F(10), F(12) };
  auto conv_at = [&](int b) {
    ConvP p{ F(b), F(b + 1), F(b + 2), F(b + 3), F(b + 4),
             F(b + 5), F(b + 6), F(b + 7), F(b + 8), F(b + 9) };
    return p;
  };
  ConvP DEC[3] = { conv_at(13), conv_at(23), conv_at(33) };
  ConvP ENC[3] = { conv_at(43), conv_at(53), conv_at(63) };
  const float* POOLW[2] = { F(73), F(74) };

  // ---- workspace bump allocator ----
  char* wsp = (char*)d_ws;
  auto alloc = [&](size_t nbytes) -> void* {
    void* p = (void*)wsp;
    wsp += (nbytes + 255) & ~(size_t)255;
    return p;
  };
  float*    Q     = (float*)alloc((size_t)cN0 * 128 * 4);
  float*    Kb    = (float*)alloc((size_t)cN0 * 128 * 4);
  float*    V     = (float*)alloc((size_t)cN0 * 128 * 4);
  float*    XR    = (float*)alloc((size_t)cN0 * 128 * 4);
  float*    AGG   = (float*)alloc((size_t)cN0 * 128 * 4);
  float*    ALPHA = (float*)alloc((size_t)cE * 4 * 4);
  unsigned* MENC  = (unsigned*)alloc((size_t)cN0 * 4 * 4);
  float*    DEN   = (float*)alloc((size_t)cN0 * 4 * 4);
  float*    CURX  = (float*)alloc((size_t)cN0 * 256 * 4);
  float*    TMPX  = (float*)alloc((size_t)cN0 * 256 * 4);
  float*    XE0   = (float*)alloc((size_t)cN0 * 128 * 4);
  float*    XE1   = (float*)alloc((size_t)cN1 * 128 * 4);
  float*    XE2   = (float*)alloc((size_t)cN2 * 128 * 4);
  bf16_t*   WPACK = (bf16_t*)alloc((size_t)256 * 128 * 2);   // max K*Nc in bf16
  float*    MEAN  = (float*)alloc(128 * 4);
  float*    VARB  = (float*)alloc(128 * 4);
  float*    WN    = (float*)alloc(256);
  float*    SCORE = (float*)alloc((size_t)cN0 * 4);
  int*      RANK  = (int*)alloc((size_t)cN0 * 4);
  int*      PERM0 = (int*)alloc((size_t)cN1 * 4);
  int*      INV0  = (int*)alloc((size_t)cN0 * 4);
  int*      PERM1 = (int*)alloc((size_t)cN2 * 4);
  int*      INV1  = (int*)alloc((size_t)cN1 * 4);
  int*      S1 = (int*)alloc((size_t)cE * 4);
  int*      D1 = (int*)alloc((size_t)cE * 4);
  int*      M1 = (int*)alloc((size_t)cE * 4);
  int*      S2 = (int*)alloc((size_t)cE * 4);
  int*      D2 = (int*)alloc((size_t)cE * 4);
  int*      M2 = (int*)alloc((size_t)cE * 4);

  // ======== encoder layer 0 (in 64, N=10000, level-0 edges) ========
  run_conv(stream, ENC[0], X0, 64, cN0, SRC0, DST0, nullptr, 4, 32, EA, WPACK,
           Q, Kb, V, XR, AGG, ALPHA, MENC, DEN, TMPX);
  run_bn_relu(stream, TMPX, BNenc_g[0], BNenc_b[0], cN0, 128, MEAN, VARB);
  k_copy_f32<<<cdiv(cN0 * 128, WG), WG, 0, stream>>>(XE0, TMPX, cN0 * 128);
  run_pool(stream, TMPX, POOLW[0], cN0, cN1, SRC0, DST0, nullptr,
           SCORE, RANK, PERM0, INV0, WN, CURX, S1, D1, M1);

  // ======== encoder layer 1 (in 128, N=5000, level-1 edges) ========
  run_conv(stream, ENC[1], CURX, 128, cN1, S1, D1, M1, 4, 32, EA, WPACK,
           Q, Kb, V, XR, AGG, ALPHA, MENC, DEN, TMPX);
  run_bn_relu(stream, TMPX, BNenc_g[1], BNenc_b[1], cN1, 128, MEAN, VARB);
  k_copy_f32<<<cdiv(cN1 * 128, WG), WG, 0, stream>>>(XE1, TMPX, cN1 * 128);
  run_pool(stream, TMPX, POOLW[1], cN1, cN2, S1, D1, M1,
           SCORE, RANK, PERM1, INV1, WN, CURX, S2, D2, M2);

  // ======== encoder layer 2 (in 128, N=2500, level-2 edges, no pool) ========
  run_conv(stream, ENC[2], CURX, 128, cN2, S2, D2, M2, 4, 32, EA, WPACK,
           Q, Kb, V, XR, AGG, ALPHA, MENC, DEN, TMPX);
  run_bn_relu(stream, TMPX, BNenc_g[2], BNenc_b[2], cN2, 128, MEAN, VARB);
  k_copy_f32<<<cdiv(cN2 * 128, WG), WG, 0, stream>>>(XE2, TMPX, cN2 * 128);
  k_copy_f32<<<cdiv(cN2 * 128, WG), WG, 0, stream>>>(CURX, TMPX, cN2 * 128);

  // ======== decoder 0 (idx=2: no unpool; concat with XE2; level-2 edges) ========
  k_copy_f32<<<cdiv(cN2 * 128, WG), WG, 0, stream>>>(TMPX, CURX, cN2 * 128);
  k_concat<<<cdiv(cN2 * 256, WG), WG, 0, stream>>>(TMPX, XE2, CURX, cN2, 128, 128);
  run_conv(stream, DEC[0], CURX, 256, cN2, S2, D2, M2, 4, 32, EA, WPACK,
           Q, Kb, V, XR, AGG, ALPHA, MENC, DEN, TMPX);
  run_bn_relu(stream, TMPX, BNdec_g[0], BNdec_b[0], cN2, 128, MEAN, VARB);
  k_copy_f32<<<cdiv(cN2 * 128, WG), WG, 0, stream>>>(CURX, TMPX, cN2 * 128);

  // ======== decoder 1 (idx=1: unpool via PERM1 to N=5000; level-1 edges) ========
  k_fill_f32<<<cdiv(cN1 * 128, WG), WG, 0, stream>>>(TMPX, 0.f, cN1 * 128);
  k_scatter_unpool<<<cdiv(cN2 * 128, WG), WG, 0, stream>>>(CURX, PERM1, TMPX, cN2, 128);
  k_concat<<<cdiv(cN1 * 256, WG), WG, 0, stream>>>(TMPX, XE1, CURX, cN1, 128, 128);
  run_conv(stream, DEC[1], CURX, 256, cN1, S1, D1, M1, 4, 32, EA, WPACK,
           Q, Kb, V, XR, AGG, ALPHA, MENC, DEN, TMPX);
  run_bn_relu(stream, TMPX, BNdec_g[1], BNdec_b[1], cN1, 128, MEAN, VARB);
  k_copy_f32<<<cdiv(cN1 * 128, WG), WG, 0, stream>>>(CURX, TMPX, cN1 * 128);

  // ======== decoder 2 (idx=0: unpool via PERM0 to N=10000; 1 head, 64 ch) ========
  k_fill_f32<<<cdiv(cN0 * 128, WG), WG, 0, stream>>>(TMPX, 0.f, cN0 * 128);
  k_scatter_unpool<<<cdiv(cN1 * 128, WG), WG, 0, stream>>>(CURX, PERM0, TMPX, cN1, 128);
  k_concat<<<cdiv(cN0 * 256, WG), WG, 0, stream>>>(TMPX, XE0, CURX, cN0, 128, 128);
  run_conv(stream, DEC[2], CURX, 256, cN0, SRC0, DST0, nullptr, 1, 64, EA, WPACK,
           Q, Kb, V, XR, AGG, ALPHA, MENC, DEN, TMPX);
  k_copy_f32<<<cdiv(cN0 * 64, WG), WG, 0, stream>>>((float*)d_out, TMPX, cN0 * 64);
}
